// DisGraphLearn_90331752169528
// MI455X (gfx1250) — matrix operand order repaired and checked by reading, not compiled, not run on hardware
//
#include <hip/hip_runtime.h>
#include <cstdint>
#include <cstddef>

// ---------------------------------------------------------------------------
// DisGraphLearn for MI455X (gfx1250, wave32, WMMA bf16 path)
//   x:[32,512,1024] f32, W:[512,512] f32  ->  out:[32,512,512] f32
// Pipeline:
//   1) fused x->bf16 convert + fp32 row stats (single HBM pass over x)
//   2) W->bf16
//   3) batched X X^T via v_wmma_f32_16x16x32_bf16, fused distance epilogue
//   4) column softmax (axis=1) -> bf16 adjacency
//   5) adj @ W^T via WMMA -> fp32 out
// ---------------------------------------------------------------------------

typedef __attribute__((ext_vector_type(16))) __bf16 v16bf;
typedef __attribute__((ext_vector_type(8)))  float  v8f;
typedef __attribute__((ext_vector_type(4)))  __bf16 v4bf;

#define DGL_B 32
#define DGL_N 512
#define DGL_F 1024
#define DGL_EPS 1e-6f

// Async global->LDS path (gfx1250 GLOBAL_LOAD_ASYNC_TO_LDS_B128).
// Probe-confirmed signature: (global int4*, local int4*, imm offset, imm cpol).
#if defined(__gfx1250__) && __has_builtin(__builtin_amdgcn_global_load_async_to_lds_b128)
#define DGL_ASYNC 1
typedef int dgl_v4i __attribute__((vector_size(16)));
typedef __attribute__((address_space(1))) dgl_v4i* dgl_gp;   // global int4*
typedef __attribute__((address_space(3))) dgl_v4i* dgl_lp;   // LDS    int4*
#else
#define DGL_ASYNC 0
#endif

__device__ __forceinline__ __bf16 dgl_f2bf(float f) {
  // round-to-nearest-even f32 -> bf16
  uint32_t u = __builtin_bit_cast(uint32_t, f);
  u += 0x7FFFu + ((u >> 16) & 1u);
  unsigned short h = (unsigned short)(u >> 16);
  return __builtin_bit_cast(__bf16, h);
}

struct DglFrag32 { uint4 lo, hi; };

__device__ __forceinline__ v16bf dgl_ld_frag(const uint4* p, int i0, int i1) {
  DglFrag32 f;
  f.lo = p[i0];
  f.hi = p[i1];
  return __builtin_bit_cast(v16bf, f);
}

__device__ __forceinline__ v8f dgl_wmma(v16bf a, v16bf b, v8f c) {
  return __builtin_amdgcn_wmma_f32_16x16x32_bf16(
      false, a, false, b, (short)0, c, false, false);
}

__device__ __forceinline__ void dgl_stage_b128(const __bf16* src, uint4* dst) {
#if DGL_ASYNC
  __builtin_amdgcn_global_load_async_to_lds_b128(
      (dgl_gp)(void*)(const void*)src, (dgl_lp)(void*)dst, 0, 0);
#else
  *dst = *(const uint4*)src;
#endif
}

__device__ __forceinline__ void dgl_stage_wait() {
#if DGL_ASYNC
#if __has_builtin(__builtin_amdgcn_s_wait_asynccnt)
  __builtin_amdgcn_s_wait_asynccnt(0);
#else
  asm volatile("s_wait_asynccnt 0" ::: "memory");
#endif
#endif
}

// ---------------------------------------------------------------------------
// 128x64-tile bf16 GEMM K-loop, 128 threads (4 waves). Each wave owns a 32x64
// strip: 2 A fragments x 4 B fragments = 8 WMMAs per 32-wide K step, with each
// B fragment register-reused by two WMMAs.
// A tile: 128 rows x 32 K, row-major bf16. BT tile: 64 rows x 32 K (rows n of
// B^T, so B[k][n] = BT[n][k]). Fragment layouts per CDNA5 ISA 7.12.2.
// ---------------------------------------------------------------------------
__device__ __forceinline__ void dgl_gemm_k_loop(
    const __bf16* __restrict__ Abase, const __bf16* __restrict__ Bbase,
    int ldA, int ldB, int Ktot, uint4* At /*[512]*/, uint4* Bt /*[256]*/,
    v8f (&acc)[2][4]) {
  const int tid  = threadIdx.x;
  const int wave = tid >> 5;
  const int lane = tid & 31;
  const int m16  = lane & 15;
  const int kh   = lane >> 4;

  for (int k0 = 0; k0 < Ktot; k0 += 32) {
    // Stage A: 128 rows * 64B = 512 uint4; B: 64 rows = 256 uint4.
#pragma unroll
    for (int c = 0; c < 4; ++c) {
      const int chunk = tid + c * 128;
      const int row = chunk >> 2, q = chunk & 3;
      dgl_stage_b128(Abase + (size_t)row * ldA + k0 + q * 8, &At[chunk]);
    }
#pragma unroll
    for (int c = 0; c < 2; ++c) {
      const int chunk = tid + c * 128;
      const int row = chunk >> 2, q = chunk & 3;
      dgl_stage_b128(Bbase + (size_t)row * ldB + k0 + q * 8, &Bt[chunk]);
    }
    dgl_stage_wait();
    __syncthreads();

    // A fragments (16x32): lane m16 = row; chunks at K {kh*8..+7, 16+kh*8..+7}
    const int ar0 = wave * 32 + m16;
    const int ar1 = ar0 + 16;
    v16bf a0 = dgl_ld_frag(At, ar0 * 4 + kh, ar0 * 4 + 2 + kh);
    v16bf a1 = dgl_ld_frag(At, ar1 * 4 + kh, ar1 * 4 + 2 + kh);

#pragma unroll
    for (int nb = 0; nb < 4; ++nb) {
      // B fragment (32x16): lane m16 = col n; chunks at K {kh*8..+7, 16+kh*8..+7}
      const int br = nb * 16 + m16;
      v16bf bm = dgl_ld_frag(Bt, br * 4 + kh, br * 4 + 2 + kh);
      acc[0][nb] = dgl_wmma(a0, bm, acc[0][nb]);
      acc[1][nb] = dgl_wmma(a1, bm, acc[1][nb]);
    }
    __syncthreads();
  }
}

// ---------------------------------------------------------------------------
// Kernel 1: fused x->bf16 + fp32 row sums (single pass over x).
// One wave32 per row of 1024 floats.
// ---------------------------------------------------------------------------
__global__ __launch_bounds__(256) void dgl_cvt_stats(
    const float* __restrict__ x, __bf16* __restrict__ xb,
    float* __restrict__ sq, float* __restrict__ s) {
  const int row  = blockIdx.x * 8 + (threadIdx.x >> 5);
  const int lane = threadIdx.x & 31;
  const float* xr = x + (size_t)row * DGL_F;
  __bf16* xbr = xb + (size_t)row * DGL_F;
  float ss = 0.0f, sl = 0.0f;
  for (int k = lane * 4; k < DGL_F; k += 128) {
    const float4 v = *(const float4*)(xr + k);
    ss += v.x * v.x + v.y * v.y + v.z * v.z + v.w * v.w;
    sl += v.x + v.y + v.z + v.w;
    v4bf o;
    o[0] = dgl_f2bf(v.x);
    o[1] = dgl_f2bf(v.y);
    o[2] = dgl_f2bf(v.z);
    o[3] = dgl_f2bf(v.w);
    *(v4bf*)(xbr + k) = o;
  }
#pragma unroll
  for (int off = 16; off > 0; off >>= 1) {
    ss += __shfl_xor(ss, off, 32);
    sl += __shfl_xor(sl, off, 32);
  }
  if (lane == 0) {
    sq[row] = ss;
    s[row]  = sl;
  }
}

// ---------------------------------------------------------------------------
// Kernel 2: W -> bf16 (4 elements/thread)
// ---------------------------------------------------------------------------
__global__ __launch_bounds__(256) void dgl_cvt_bf16(
    const float* __restrict__ in, __bf16* __restrict__ out, int n4) {
  const int t = blockIdx.x * 256 + threadIdx.x;
  if (t >= n4) return;
  const float4 v = ((const float4*)in)[t];
  v4bf o;
  o[0] = dgl_f2bf(v.x);
  o[1] = dgl_f2bf(v.y);
  o[2] = dgl_f2bf(v.z);
  o[3] = dgl_f2bf(v.w);
  ((v4bf*)out)[t] = o;
}

// ---------------------------------------------------------------------------
// Kernel 3: batched X X^T via WMMA + fused pairwise-distance epilogue
//   dist = sqrt(max(sq_i + sq_j - 2*dot + 2*eps*(s_i - s_j) + F*eps^2, 0))
//   diagonal forced to exact zero (reference skips i==j)
// ---------------------------------------------------------------------------
__global__ __launch_bounds__(128) void dgl_gemm_dist(
    const __bf16* __restrict__ xb, const float* __restrict__ sq,
    const float* __restrict__ s, float* __restrict__ dist) {
  __shared__ uint4 At[512];
  __shared__ uint4 Bt[256];
  const int b     = blockIdx.z;
  const int iBase = blockIdx.y * 128;
  const int jBase = blockIdx.x * 64;
  const __bf16* Ab = xb + ((size_t)b * DGL_N + iBase) * DGL_F;
  const __bf16* Bb = xb + ((size_t)b * DGL_N + jBase) * DGL_F;

  v8f acc[2][4] = {};
  dgl_gemm_k_loop(Ab, Bb, DGL_F, DGL_F, DGL_F, At, Bt, acc);

  const int lane = threadIdx.x & 31, wave = threadIdx.x >> 5;
  const int m16 = lane & 15, kh = lane >> 4;
  const float* sqb = sq + b * DGL_N;
  const float* sb  = s  + b * DGL_N;
  const float cst = (float)DGL_F * DGL_EPS * DGL_EPS;
#pragma unroll
  for (int ri = 0; ri < 2; ++ri) {
#pragma unroll
    for (int nb = 0; nb < 4; ++nb) {
      const int j = jBase + nb * 16 + m16;     // C/D layout: N = lane&15
      const float sqj = sqb[j], sj = sb[j];
#pragma unroll
      for (int r = 0; r < 8; ++r) {
        const int i = iBase + wave * 32 + ri * 16 + kh * 8 + r;  // M = r(+8)
        float d2 = sqb[i] + sqj - 2.0f * acc[ri][nb][r]
                 + 2.0f * DGL_EPS * (sb[i] - sj) + cst;
        float dv = sqrtf(fmaxf(d2, 0.0f));
        if (i == j) dv = 0.0f;
        dist[((size_t)b * DGL_N + i) * DGL_N + j] = dv;
      }
    }
  }
}

// ---------------------------------------------------------------------------
// Kernel 4: softmax over axis=1 (over i, per column j); output bf16 adj.
// One thread per (b, j): consecutive lanes -> consecutive j, so every access
// across the wave is coalesced; working set is L2-resident.
// ---------------------------------------------------------------------------
__global__ __launch_bounds__(256) void dgl_softmax_col(
    const float* __restrict__ dist, __bf16* __restrict__ adjb) {
  const int idx = blockIdx.x * 256 + threadIdx.x;  // b*N + j
  const int b = idx >> 9;
  const int j = idx & (DGL_N - 1);
  const float* col = dist + (size_t)b * DGL_N * DGL_N + j;
  float mx = 0.0f;               // distances >= 0 (diag exactly 0)
  for (int i = 0; i < DGL_N; ++i) mx = fmaxf(mx, col[(size_t)i * DGL_N]);
  float sum = 0.0f;
  for (int i = 0; i < DGL_N; ++i) sum += __expf(col[(size_t)i * DGL_N] - mx);
  const float inv = 1.0f / sum;
  for (int i = 0; i < DGL_N; ++i) {
    const float p = __expf(col[(size_t)i * DGL_N] - mx) * inv;
    adjb[((size_t)b * DGL_N + i) * DGL_N + j] = dgl_f2bf(p);
  }
}

// ---------------------------------------------------------------------------
// Kernel 5: out[b,i,k] = sum_j adj[b,i,j] * W[k,j]  (adj @ W^T) via WMMA.
// BT tile rows n=k are rows of W (row-major), K dimension = j.
// ---------------------------------------------------------------------------
__global__ __launch_bounds__(128) void dgl_gemm_out(
    const __bf16* __restrict__ adjb, const __bf16* __restrict__ Wb,
    float* __restrict__ out) {
  __shared__ uint4 At[512];
  __shared__ uint4 Bt[256];
  const int b     = blockIdx.z;
  const int iBase = blockIdx.y * 128;
  const int kBase = blockIdx.x * 64;
  const __bf16* Ab = adjb + ((size_t)b * DGL_N + iBase) * DGL_N;
  const __bf16* Bb = Wb + (size_t)kBase * DGL_N;

  v8f acc[2][4] = {};
  dgl_gemm_k_loop(Ab, Bb, DGL_N, DGL_N, DGL_N, At, Bt, acc);

  const int lane = threadIdx.x & 31, wave = threadIdx.x >> 5;
  const int m16 = lane & 15, kh = lane >> 4;
#pragma unroll
  for (int ri = 0; ri < 2; ++ri) {
#pragma unroll
    for (int nb = 0; nb < 4; ++nb) {
      const int kc = kBase + nb * 16 + m16;
#pragma unroll
      for (int r = 0; r < 8; ++r) {
        const int i = iBase + wave * 32 + ri * 16 + kh * 8 + r;
        out[((size_t)b * DGL_N + i) * DGL_N + kc] = acc[ri][nb][r];
      }
    }
  }
}

// ---------------------------------------------------------------------------
// Launch
// ---------------------------------------------------------------------------
extern "C" void kernel_launch(void* const* d_in, const int* in_sizes, int n_in,
                              void* d_out, int out_size, void* d_ws, size_t ws_size,
                              hipStream_t stream) {
  (void)in_sizes; (void)n_in; (void)out_size; (void)ws_size;
  const float* x = (const float*)d_in[0];   // [32,512,1024] f32
  const float* W = (const float*)d_in[1];   // [512,512]     f32
  float* out = (float*)d_out;               // [32,512,512]  f32

  char* ws = (char*)d_ws;
  size_t off = 0;
  auto carve = [&](size_t bytes) -> char* {
    char* p = ws + off;
    off = (off + bytes + 255) & ~(size_t)255;
    return p;
  };
  __bf16* xb   = (__bf16*)carve((size_t)DGL_B * DGL_N * DGL_F * 2);  // 33.5 MB
  __bf16* Wb   = (__bf16*)carve((size_t)DGL_N * DGL_N * 2);          // 0.5 MB
  float*  sq   = (float*) carve((size_t)DGL_B * DGL_N * 4);
  float*  s    = (float*) carve((size_t)DGL_B * DGL_N * 4);
  float*  dist = (float*) carve((size_t)DGL_B * DGL_N * DGL_N * 4);  // 33.5 MB
  __bf16* adjb = (__bf16*)carve((size_t)DGL_B * DGL_N * DGL_N * 2);  // 16.8 MB

  // 1) fused x->bf16 + row stats (single HBM pass over x)
  dgl_cvt_stats<<<(DGL_B * DGL_N) / 8, 256, 0, stream>>>(x, xb, sq, s);

  // 2) W -> bf16
  {
    const int n4 = DGL_N * DGL_N / 4;
    dgl_cvt_bf16<<<(n4 + 255) / 256, 256, 0, stream>>>(W, Wb, n4);
  }

  // 3) X X^T + pairwise-distance epilogue
  {
    dim3 grid(DGL_N / 64, DGL_N / 128, DGL_B);
    dgl_gemm_dist<<<grid, 128, 0, stream>>>(xb, sq, s, dist);
  }

  // 4) column softmax -> bf16 adjacency
  dgl_softmax_col<<<(DGL_B * DGL_N) / 256, 256, 0, stream>>>(dist, adjb);

  // 5) adj @ W^T
  {
    dim3 grid(DGL_N / 64, DGL_N / 128, DGL_B);
    dgl_gemm_out<<<grid, 128, 0, stream>>>(adjb, Wb, out);
  }
}
